// FFTBlock_89833535963935
// MI455X (gfx1250) — matrix operand order repaired and checked by reading, not compiled, not run on hardware
//
#include <hip/hip_runtime.h>
#include <hip/hip_bf16.h>
#include <math.h>

// Problem constants (match reference)
#define LL 2048
#define BB 2
#define CC 1024
#define HH 16
#define HD 64
#define FFC 4096
#define KW 9
#define RR (LL*BB)   /* 4096 rows in (L,B) flattening */
#define C3 (3*CC)    /* 3072 */

typedef __attribute__((ext_vector_type(16))) _Float16 v16h;
typedef __attribute__((ext_vector_type(8)))  float    v8f;

union AFrag { v16h v; unsigned u[8]; _Float16 h[16]; int4 q[2]; };

// ---- CDNA5 async global->LDS staging (guarded: falls back to ld/st copies) ----
#if defined(__AMDGCN__) && __has_builtin(__builtin_amdgcn_global_load_async_to_lds_b128)
#define HAVE_ASYNC_LDS 1
#else
#define HAVE_ASYNC_LDS 0
#endif

#if HAVE_ASYNC_LDS
typedef int vi4 __attribute__((vector_size(16)));               // native vector, not HIP_vector_type
typedef __attribute__((address_space(1))) vi4 gas_vi4;          // global
typedef __attribute__((address_space(3))) vi4 las_vi4;          // LDS
__device__ __forceinline__ void async_cp16(void* lds, const void* g) {
  __builtin_amdgcn_global_load_async_to_lds_b128(
      (gas_vi4*)g, (las_vi4*)lds, 0, 0);
}
#endif

__device__ __forceinline__ void wait_async_lds() {
#if HAVE_ASYNC_LDS
#if __has_builtin(__builtin_amdgcn_s_wait_asynccnt)
  __builtin_amdgcn_s_wait_asynccnt(0);
#else
  asm volatile("s_wait_asynccnt 0x0" ::: "memory");
#endif
#endif
}

// ---- wave32 16-lane-group reductions (rows of a C-fragment live across 16 lanes)
__device__ __forceinline__ float redmax16(float v) {
#pragma unroll
  for (int o = 8; o > 0; o >>= 1) v = fmaxf(v, __shfl_xor(v, o, 32));
  return v;
}
__device__ __forceinline__ float redsum16(float v) {
#pragma unroll
  for (int o = 8; o > 0; o >>= 1) v += __shfl_xor(v, o, 32);
  return v;
}

// ---------------- Tiled WMMA GEMM:  Out[M,N](f32) = A[M,K](f16) @ W[N,K](f16)^T + bias[N]
// B operand staged COLUMN-major (== W row-major): every fragment load is two ds_load_b128.
// Double-buffered LDS; next tile staged (async if available) while current computes.
#define TM 128
#define TN 128
#define TK 32
#define LDA 40   /* 80B rows: 16B aligned, conflict-reducing pad */

__global__ __launch_bounds__(256) void gemm_f16f32(
    const _Float16* __restrict__ A, const _Float16* __restrict__ W,
    const float* __restrict__ bias, float* __restrict__ Out,
    int M, int N, int K) {
  __shared__ __align__(16) _Float16 As[2][TM][LDA];
  __shared__ __align__(16) _Float16 Bs[2][TN][LDA];
  const int tm = blockIdx.y * TM, tn = blockIdx.x * TN;
  const int tid = threadIdx.x;
  const int wave = tid >> 5, lane = tid & 31;
  const int l16 = lane & 15, hi = lane >> 4;
  const int r = tid >> 1, c = (tid & 1) * 16;
  const _Float16* aptr = A + (size_t)(tm + r) * K + c;
  const _Float16* bptr = W + (size_t)(tn + r) * K + c;

  auto stage = [&](int koff, int buf) {
#if HAVE_ASYNC_LDS
    async_cp16(&As[buf][r][c],     aptr + koff);
    async_cp16(&As[buf][r][c + 8], aptr + koff + 8);
    async_cp16(&Bs[buf][r][c],     bptr + koff);
    async_cp16(&Bs[buf][r][c + 8], bptr + koff + 8);
#else
    int4 t0 = *(const int4*)(aptr + koff);
    int4 t1 = *(const int4*)(aptr + koff + 8);
    int4 t2 = *(const int4*)(bptr + koff);
    int4 t3 = *(const int4*)(bptr + koff + 8);
    *(int4*)&As[buf][r][c]     = t0;
    *(int4*)&As[buf][r][c + 8] = t1;
    *(int4*)&Bs[buf][r][c]     = t2;
    *(int4*)&Bs[buf][r][c + 8] = t3;
#endif
  };

  v8f acc[8] = {};
  int p = 0;
  stage(0, 0);
  for (int k0 = 0; k0 < K; k0 += TK) {
    wait_async_lds();
    __syncthreads();
    if (k0 + TK < K) stage(k0 + TK, p ^ 1);
    AFrag a;
    a.q[0] = *(const int4*)&As[p][wave * 16 + l16][hi * 8];
    a.q[1] = *(const int4*)&As[p][wave * 16 + l16][16 + hi * 8];
    AFrag bf[8];
#pragma unroll
    for (int cf = 0; cf < 8; ++cf) {
      bf[cf].q[0] = *(const int4*)&Bs[p][cf * 16 + l16][hi * 8];
      bf[cf].q[1] = *(const int4*)&Bs[p][cf * 16 + l16][16 + hi * 8];
    }
#pragma unroll
    for (int cf = 0; cf < 8; ++cf)
      acc[cf] = __builtin_amdgcn_wmma_f32_16x16x32_f16(
          false, a.v, false, bf[cf].v, (short)0, acc[cf], false, false);
    p ^= 1;
  }
#pragma unroll
  for (int cf = 0; cf < 8; ++cf) {
    int gn = tn + cf * 16 + l16;
    float bv = bias ? bias[gn] : 0.f;
#pragma unroll
    for (int i = 0; i < 8; ++i) {
      int gm = tm + wave * 16 + i + hi * 8;
      Out[(size_t)gm * N + gn] = acc[cf][i] + bv;
    }
  }
}

// ---------------- Conv1 as implicit GEMM over (tap,channel)-K, ReLU, f16 out
// rows: g = b*L + l ; A[g, c] at tap kt = x1h[((l+kt-4)*B + b)*C + c] (zero OOB)
// weights pre-shuffled to w1cm[kt][f][c] so B tiles are contiguous (column-major B).
__global__ __launch_bounds__(256) void conv1_gemm(
    const _Float16* __restrict__ x1h, const _Float16* __restrict__ w1cm,
    const float* __restrict__ bias, _Float16* __restrict__ hout) {
  __shared__ __align__(16) _Float16 As[2][TM][LDA];
  __shared__ __align__(16) _Float16 Bs[2][TN][LDA];
  const int tm = blockIdx.y * TM, tn = blockIdx.x * TN;
  const int tid = threadIdx.x;
  const int wave = tid >> 5, lane = tid & 31;
  const int l16 = lane & 15, hi = lane >> 4;
  const int r = tid >> 1, c = (tid & 1) * 16;
  const int g = tm + r, b = g >> 11, l = g & (LL - 1);

  auto stage = [&](int step, int buf) {   // step = kt*32 + (c0/32)
    int kt = step >> 5, c0 = (step & 31) * TK;
    int lp = l + kt - KW / 2;
    const _Float16* bsrc = w1cm + (size_t)(kt * FFC + tn + r) * CC + c0 + c;
    if (lp >= 0 && lp < LL) {
      const _Float16* asrc = x1h + (size_t)(lp * BB + b) * CC + c0 + c;
#if HAVE_ASYNC_LDS
      async_cp16(&As[buf][r][c],     asrc);
      async_cp16(&As[buf][r][c + 8], asrc + 8);
#else
      int4 t0 = *(const int4*)asrc;
      int4 t1 = *(const int4*)(asrc + 8);
      *(int4*)&As[buf][r][c]     = t0;
      *(int4*)&As[buf][r][c + 8] = t1;
#endif
    } else {
      int4 z = make_int4(0, 0, 0, 0);
      *(int4*)&As[buf][r][c] = z;
      *(int4*)&As[buf][r][c + 8] = z;
    }
#if HAVE_ASYNC_LDS
    async_cp16(&Bs[buf][r][c],     bsrc);
    async_cp16(&Bs[buf][r][c + 8], bsrc + 8);
#else
    int4 t2 = *(const int4*)bsrc;
    int4 t3 = *(const int4*)(bsrc + 8);
    *(int4*)&Bs[buf][r][c]     = t2;
    *(int4*)&Bs[buf][r][c + 8] = t3;
#endif
  };

  v8f acc[8] = {};
  const int NSTEP = KW * (CC / TK);   // 9 * 32 = 288
  int p = 0;
  stage(0, 0);
  for (int s = 0; s < NSTEP; ++s) {
    wait_async_lds();
    __syncthreads();
    if (s + 1 < NSTEP) stage(s + 1, p ^ 1);
    AFrag a;
    a.q[0] = *(const int4*)&As[p][wave * 16 + l16][hi * 8];
    a.q[1] = *(const int4*)&As[p][wave * 16 + l16][16 + hi * 8];
    AFrag bf[8];
#pragma unroll
    for (int cf = 0; cf < 8; ++cf) {
      bf[cf].q[0] = *(const int4*)&Bs[p][cf * 16 + l16][hi * 8];
      bf[cf].q[1] = *(const int4*)&Bs[p][cf * 16 + l16][16 + hi * 8];
    }
#pragma unroll
    for (int cf = 0; cf < 8; ++cf)
      acc[cf] = __builtin_amdgcn_wmma_f32_16x16x32_f16(
          false, a.v, false, bf[cf].v, (short)0, acc[cf], false, false);
    p ^= 1;
  }
#pragma unroll
  for (int cf = 0; cf < 8; ++cf) {
    int gn = tn + cf * 16 + l16;
    float bv = bias[gn];
#pragma unroll
    for (int i = 0; i < 8; ++i) {
      int gm = tm + wave * 16 + i + hi * 8;
      hout[(size_t)gm * FFC + gn] = (_Float16)fmaxf(acc[cf][i] + bv, 0.f);
    }
  }
}

// ---------------- Flash attention: 1 block per (b,h, 64-query tile); 4 waves x 16 rows
#define LDK 72   /* K tile row stride (64 + 8 pad), 144B rows */
__global__ __launch_bounds__(128) void flash_attn(
    const _Float16* __restrict__ qh, const _Float16* __restrict__ kh,
    const _Float16* __restrict__ vh, const unsigned char* __restrict__ mask,
    _Float16* __restrict__ aoh) {
  __shared__ __align__(16) _Float16 Ks[32][LDK];     // [key][d] == column-major B for Q@K^T
  __shared__ __align__(16) _Float16 Vt[64][LDA];     // [d][key] == column-major B for P@V
  __shared__ __align__(16) _Float16 Ps[4][16][LDA];  // per-wave P tile, row-major
  const int bh = blockIdx.y;
  const int b = bh / HH, h = bh % HH;
  const int tid = threadIdx.x, wave = tid >> 5, lane = tid & 31;
  const int l16 = lane & 15, hi = lane >> 4;
  const int q0 = blockIdx.x * 64 + wave * 16;
  const size_t head_base = (size_t)bh * LL * HD;

  AFrag qa[2]; // resident Q rows: 16x64 as two 16x32 A-frags
  {
    const _Float16* qrow = qh + head_base + (size_t)(q0 + l16) * HD;
    qa[0].q[0] = *(const int4*)(qrow + hi * 8);
    qa[0].q[1] = *(const int4*)(qrow + 16 + hi * 8);
    qa[1].q[0] = *(const int4*)(qrow + 32 + hi * 8);
    qa[1].q[1] = *(const int4*)(qrow + 48 + hi * 8);
  }
  v8f o[4] = {};
  float mrun[8], lrun[8];
#pragma unroll
  for (int i = 0; i < 8; ++i) { mrun[i] = -INFINITY; lrun[i] = 0.f; }

  for (int kb = 0; kb < LL / 32; ++kb) {
    __syncthreads();
    { // stage K tile (natural layout) + V tile transposed
      int rk = tid >> 2, ck = (tid & 3) * 16;
      const _Float16* ksrc = kh + head_base + (size_t)(kb * 32 + rk) * HD + ck;
      int key = tid & 31, d0 = (tid >> 5) * 16;
      const _Float16* vsrc = vh + head_base + (size_t)(kb * 32 + key) * HD + d0;
      int4 t0 = *(const int4*)ksrc;
      int4 t1 = *(const int4*)(ksrc + 8);
      AFrag vload;
      vload.q[0] = *(const int4*)vsrc;
      vload.q[1] = *(const int4*)(vsrc + 8);
      *(int4*)&Ks[rk][ck]     = t0;
      *(int4*)&Ks[rk][ck + 8] = t1;
#pragma unroll
      for (int i = 0; i < 16; ++i) Vt[d0 + i][key] = vload.h[i];
    }
    __syncthreads();
    // S = (Q K^T) for 32 keys: two 16x16 C-frags, K-dim = hd = 64
    v8f s0 = {}, s1 = {};
#pragma unroll
    for (int f = 0; f < 2; ++f) {
      AFrag bk0, bk1;
      bk0.q[0] = *(const int4*)&Ks[l16][f * 32 + hi * 8];
      bk0.q[1] = *(const int4*)&Ks[l16][f * 32 + 16 + hi * 8];
      bk1.q[0] = *(const int4*)&Ks[16 + l16][f * 32 + hi * 8];
      bk1.q[1] = *(const int4*)&Ks[16 + l16][f * 32 + 16 + hi * 8];
      s0 = __builtin_amdgcn_wmma_f32_16x16x32_f16(false, qa[f].v, false, bk0.v, (short)0, s0, false, false);
      s1 = __builtin_amdgcn_wmma_f32_16x16x32_f16(false, qa[f].v, false, bk1.v, (short)0, s1, false, false);
    }
    const float sc = 0.125f; // 1/sqrt(64)
    const bool msk0 = mask[b * LL + kb * 32 + l16] != 0;
    const bool msk1 = mask[b * LL + kb * 32 + 16 + l16] != 0;
#pragma unroll
    for (int i = 0; i < 8; ++i) {
      float a0 = msk0 ? -1e30f : s0[i] * sc;
      float a1 = msk1 ? -1e30f : s1[i] * sc;
      float mloc = redmax16(fmaxf(a0, a1));
      float mnew = fmaxf(mrun[i], mloc);
      float corr = __expf(mrun[i] - mnew);
      float p0 = __expf(a0 - mnew);
      float p1 = __expf(a1 - mnew);
      lrun[i] = lrun[i] * corr + redsum16(p0 + p1);
      mrun[i] = mnew;
#pragma unroll
      for (int cf = 0; cf < 4; ++cf) o[cf][i] *= corr;
      Ps[wave][i + hi * 8][l16]      = (_Float16)p0;
      Ps[wave][i + hi * 8][16 + l16] = (_Float16)p1;
    }
    __syncthreads();
    AFrag pa;
    pa.q[0] = *(const int4*)&Ps[wave][l16][hi * 8];
    pa.q[1] = *(const int4*)&Ps[wave][l16][16 + hi * 8];
#pragma unroll
    for (int cf = 0; cf < 4; ++cf) {
      AFrag vb;
      vb.q[0] = *(const int4*)&Vt[cf * 16 + l16][hi * 8];
      vb.q[1] = *(const int4*)&Vt[cf * 16 + l16][16 + hi * 8];
      o[cf] = __builtin_amdgcn_wmma_f32_16x16x32_f16(
          false, pa.v, false, vb.v, (short)0, o[cf], false, false);
    }
  }
  // epilogue: normalize rows and scatter to (L,B,C) layout in f16
#pragma unroll
  for (int i = 0; i < 8; ++i) {
    float rinv = 1.f / lrun[i];
    int l = q0 + i + hi * 8;
#pragma unroll
    for (int cf = 0; cf < 4; ++cf) {
      int d = cf * 16 + l16;
      aoh[(size_t)(l * BB + b) * CC + h * HD + d] = (_Float16)(o[cf][i] * rinv);
    }
  }
}

// ---------------- Residual add + LayerNorm (row = one block of 256 threads)
__global__ __launch_bounds__(256) void add_ln(
    const float* __restrict__ A, const float* __restrict__ Bp,
    const float* __restrict__ w, const float* __restrict__ bb,
    float* __restrict__ outf, _Float16* __restrict__ outh, int swap) {
  const int row = blockIdx.x;
  int rb = row;
  if (swap) { int l = row >> 1, b = row & 1; rb = b * LL + l; }  // (l*B+b) -> (b*L+l)
  const float* ap = A + (size_t)row * CC;
  const float* bp = Bp + (size_t)rb * CC;
  const int tid = threadIdx.x;
  float v[4];
  float s = 0.f;
#pragma unroll
  for (int i = 0; i < 4; ++i) { v[i] = ap[tid + i * 256] + bp[tid + i * 256]; s += v[i]; }
  __shared__ float red[256];
  red[tid] = s; __syncthreads();
  for (int st = 128; st > 0; st >>= 1) { if (tid < st) red[tid] += red[tid + st]; __syncthreads(); }
  float mean = red[0] / CC;
  __syncthreads();
  float s2 = 0.f;
#pragma unroll
  for (int i = 0; i < 4; ++i) { float d = v[i] - mean; s2 += d * d; }
  red[tid] = s2; __syncthreads();
  for (int st = 128; st > 0; st >>= 1) { if (tid < st) red[tid] += red[tid + st]; __syncthreads(); }
  float rstd = rsqrtf(red[0] / CC + 1e-5f);
#pragma unroll
  for (int i = 0; i < 4; ++i) {
    int cc = tid + i * 256;
    float y = (v[i] - mean) * rstd * w[cc] + bb[cc];
    if (outf) outf[(size_t)row * CC + cc] = y;
    if (outh) outh[(size_t)row * CC + cc] = (_Float16)y;
  }
}

// ---------------- small conversion kernels
__global__ void f32_to_f16_k(const float* __restrict__ in, _Float16* __restrict__ out, int n) {
  int i = blockIdx.x * 256 + threadIdx.x;
  if (i < n) out[i] = (_Float16)in[i];
}
// w1cm[kt][f][c] = conv1_w[f][c][kt]
__global__ void conv1w_tr_k(const float* __restrict__ w, _Float16* __restrict__ out) {
  int idx = blockIdx.x * 256 + threadIdx.x; // KW*FFC*CC
  int kt = idx / (FFC * CC);
  int rem = idx - kt * (FFC * CC);
  int f = rem / CC, cc = rem - f * CC;
  out[idx] = (_Float16)w[(size_t)f * CC * KW + cc * KW + kt];
}
// qkv f32 (rows l*B+b, 3C) -> q/k/v f16 (B,H,L,hd)
__global__ void split_qkv_k(const float* __restrict__ qkv,
                            _Float16* __restrict__ qh, _Float16* __restrict__ kh,
                            _Float16* __restrict__ vh) {
  int idx = blockIdx.x * 256 + threadIdx.x; // RR*CC
  int r = idx >> 10, cc = idx & 1023;
  int l = r >> 1, b = r & 1;
  int h = cc >> 6, d = cc & 63;
  size_t dst = (((size_t)(b * HH + h)) * LL + l) * HD + d;
  const float* src = qkv + (size_t)r * C3 + cc;
  qh[dst] = (_Float16)src[0];
  kh[dst] = (_Float16)src[CC];
  vh[dst] = (_Float16)src[2 * CC];
}

extern "C" void kernel_launch(void* const* d_in, const int* in_sizes, int n_in,
                              void* d_out, int out_size, void* d_ws, size_t ws_size,
                              hipStream_t stream) {
  (void)in_sizes; (void)n_in; (void)out_size; (void)ws_size;
  const float* x      = (const float*)d_in[0];
  const unsigned char* mask = (const unsigned char*)d_in[1];
  const float* in_w   = (const float*)d_in[2];
  const float* in_b   = (const float*)d_in[3];
  const float* out_w  = (const float*)d_in[4];
  const float* out_b  = (const float*)d_in[5];
  const float* c1w    = (const float*)d_in[6];
  const float* c1b    = (const float*)d_in[7];
  const float* c2w    = (const float*)d_in[8];
  const float* c2b    = (const float*)d_in[9];
  const float* n1w    = (const float*)d_in[10];
  const float* n1b    = (const float*)d_in[11];
  const float* n2w    = (const float*)d_in[12];
  const float* n2b    = (const float*)d_in[13];

  char* ws = (char*)d_ws;
  size_t off = 0;
  auto alloc = [&](size_t bytes) { size_t o = off; off += (bytes + 255) & ~(size_t)255; return o; };
  _Float16* xh    = (_Float16*)(ws + alloc((size_t)RR * CC * 2));
  _Float16* wqkvh = (_Float16*)(ws + alloc((size_t)C3 * CC * 2));   // in_proj_w as f16 (N x K)
  float*    qkvf  = (float*)   (ws + alloc((size_t)RR * C3 * 4));
  _Float16* qhp   = (_Float16*)(ws + alloc((size_t)RR * CC * 2));
  _Float16* khp   = (_Float16*)(ws + alloc((size_t)RR * CC * 2));
  _Float16* vhp   = (_Float16*)(ws + alloc((size_t)RR * CC * 2));
  _Float16* aoh   = (_Float16*)(ws + alloc((size_t)RR * CC * 2));
  _Float16* wouth = (_Float16*)(ws + alloc((size_t)CC * CC * 2));   // out_proj_w as f16 (N x K)
  float*    projf = (float*)   (ws + alloc((size_t)RR * CC * 4));
  float*    x1f   = (float*)   (ws + alloc((size_t)RR * CC * 4));
  _Float16* x1h   = (_Float16*)(ws + alloc((size_t)RR * CC * 2));
  _Float16* w1cm  = (_Float16*)(ws + alloc((size_t)KW * FFC * CC * 2));
  _Float16* h1h   = (_Float16*)(ws + alloc((size_t)RR * FFC * 2));
  _Float16* w2h   = (_Float16*)(ws + alloc((size_t)CC * FFC * 2));  // conv2_w as f16 (N=C x K=FF)
  float*    c2f   = (float*)   (ws + alloc((size_t)RR * CC * 4));

  // --- conversions (all plain copies except conv1 tap shuffle) ---
  f32_to_f16_k<<<(RR * CC) / 256, 256, 0, stream>>>(x, xh, RR * CC);
  f32_to_f16_k<<<(C3 * CC) / 256, 256, 0, stream>>>(in_w, wqkvh, C3 * CC);
  f32_to_f16_k<<<(CC * CC) / 256, 256, 0, stream>>>(out_w, wouth, CC * CC);
  conv1w_tr_k<<<(KW * FFC * CC) / 256, 256, 0, stream>>>(c1w, w1cm);
  f32_to_f16_k<<<(CC * FFC) / 256, 256, 0, stream>>>(c2w, w2h, CC * FFC);

  // --- attention path ---
  gemm_f16f32<<<dim3(C3 / TN, RR / TM), 256, 0, stream>>>(xh, wqkvh, in_b, qkvf, RR, C3, CC);
  split_qkv_k<<<(RR * CC) / 256, 256, 0, stream>>>(qkvf, qhp, khp, vhp);
  flash_attn<<<dim3(LL / 64, BB * HH), 128, 0, stream>>>(qhp, khp, vhp, mask, aoh);
  gemm_f16f32<<<dim3(CC / TN, RR / TM), 256, 0, stream>>>(aoh, wouth, out_b, projf, RR, CC, CC);
  add_ln<<<RR, 256, 0, stream>>>(x, projf, n1w, n1b, x1f, x1h, 0);

  // --- conv FFN path ---
  conv1_gemm<<<dim3(FFC / TN, RR / TM), 256, 0, stream>>>(x1h, w1cm, c1b, h1h);
  gemm_f16f32<<<dim3(CC / TN, RR / TM), 256, 0, stream>>>(h1h, w2h, c2b, c2f, RR, CC, FFC);
  add_ln<<<RR, 256, 0, stream>>>(x1f, c2f, n2w, n2b, (float*)d_out, (_Float16*)nullptr, 1);
}
